// MultiViewSubspaceClustering_86088324481662
// MI455X (gfx1250) — matrix operand order repaired and checked by reading, not compiled, not run on hardware
//
#include <hip/hip_runtime.h>
#include <hip/hip_bf16.h>

// ---------------------------------------------------------------------------
// MI455X (gfx1250, CDNA5) implementation of the multi-view subspace
// clustering forward pass.  All dense matmuls run through
// v_wmma_f32_16x16x32_bf16 (fp32 accumulate), operands converted to bf16
// once.  GEMM uses a double-buffered LDS pipeline fed by
// GLOBAL_LOAD_ASYNC_TO_LDS_B128, synchronized with s_wait_asynccnt.
// Epilogues / kNN / reductions are streaming kernels.
// ---------------------------------------------------------------------------

#define N_   4096
#define D_   2000
#define HID_ 256
#define ZG_  64
#define ZS_  64
#define R_   64
#define KNN_ 21      // k+1 = 20+1 (top-k includes self)

typedef __attribute__((ext_vector_type(16))) __bf16         v16bf;
typedef __attribute__((ext_vector_type(8)))  float          v8f;
typedef __attribute__((ext_vector_type(8)))  unsigned short v8us;

struct V8Pair { v8us lo; v8us hi; };
union FragCvt { V8Pair p; v16bf f; };

static __device__ __forceinline__ unsigned short f2bf(float f) {
  union { float f; unsigned u; } v; v.f = f;
  unsigned r = v.u + 0x7FFFu + ((v.u >> 16) & 1u);   // round-to-nearest-even
  return (unsigned short)(r >> 16);
}

// ---- CDNA5 async global->LDS copy (guarded; falls back to sync staging) ---
#if defined(__has_builtin)
#  if __has_builtin(__builtin_amdgcn_global_load_async_to_lds_b128)
#    define HAS_ASYNC_LDS 1
#  endif
#endif
#ifndef HAS_ASYNC_LDS
#  define HAS_ASYNC_LDS 0
#endif

#if HAS_ASYNC_LDS
// Builtin prototype (from clang diagnostics): pointee is a gcc-style
// vector_size(16) int4; param 1 in global (addrspace 1), param 2 in LDS
// (addrspace 3).  Generic -> addrspace casts via integer round-trip (the LDS
// offset lives in the low 32 bits of a generic LDS pointer on AMDGPU).
typedef int gv4i __attribute__((vector_size(16)));
#define AS1C(p) ((__attribute__((address_space(1))) gv4i*)(unsigned long long)(const void*)(p))
#define AS3C(p) ((__attribute__((address_space(3))) gv4i*)(unsigned)(unsigned long long)(void*)(p))
static __device__ __forceinline__ void wait_async_lds() {
#if __has_builtin(__builtin_amdgcn_s_wait_asynccnt)
  __builtin_amdgcn_s_wait_asynccnt(0);
#else
  asm volatile("s_wait_asynccnt 0" ::: "memory");
#endif
}
#else
static __device__ __forceinline__ void wait_async_lds() {}
#endif

// ---------------------------------------------------------------------------
// WMMA GEMM: C(MxN) = epi(A @ B + bias)
//   A: bf16, row-major MxK   (TA=1: logical A(m,k) = A[k*lda + m])
//   B: bf16, row-major KxN   (TB=1: B given NxK row-major, i.e. B(k,n)=B[n*ldb+k])
//   EPI: 0 = linear (+bias if non-null), 1 = relu(+bias)
//   Writes fp32 Cf and/or bf16 Cbf (either may be null).
// Block tile 128x128, K-step 32, 8 waves; wave tile 32x64 (2x4 WMMA tiles).
// Double-buffered LDS; async global->LDS staging when available.
// ---------------------------------------------------------------------------
#define BM 128
#define BN 128
#define BK 32
#define LDST 40   // padded LDS row stride in ushorts (80B, 16B-aligned)

template<int TA, int TB, int EPI>
__global__ __launch_bounds__(256)
void gemm_bf16_wmma(const unsigned short* __restrict__ A,
                    const unsigned short* __restrict__ B,
                    const float* __restrict__ bias,
                    float* __restrict__ Cf,
                    unsigned short* __restrict__ Cbf,
                    int M, int N, int K, int lda, int ldb, int ldc)
{
  __shared__ __align__(16) unsigned short As[2][BM * LDST];
  __shared__ __align__(16) unsigned short Bs[2][BN * LDST];

  const int tid  = threadIdx.x;
  const int lane = tid & 31;
  const int wave = tid >> 5;
  const int wr   = wave >> 1;   // 0..3
  const int wc   = wave & 1;    // 0..1
  const int bm   = blockIdx.y * BM;
  const int bn   = blockIdx.x * BN;

  const int lrow = tid >> 1;         // 0..127
  const int lk   = (tid & 1) * 16;   // 0 or 16

  // ---- tile staging (async fast path; clamped branch-free edge path) ------
  auto stage = [&](int kk, int buf) {
    unsigned short* Asb = &As[buf][0];
    unsigned short* Bsb = &Bs[buf][0];
    // A tile: Asb[row][k] = A(bm+row, kk+k)
    {
      const int gr = bm + lrow;
      if (!TA && gr < M && (kk + lk + 16) <= K) {
        const unsigned short* gp = &A[(size_t)gr * lda + kk + lk];
#if HAS_ASYNC_LDS
        __builtin_amdgcn_global_load_async_to_lds_b128(AS1C(gp),     AS3C(&Asb[lrow * LDST + lk]),     0, 0);
        __builtin_amdgcn_global_load_async_to_lds_b128(AS1C(gp + 8), AS3C(&Asb[lrow * LDST + lk + 8]), 0, 0);
#else
        const v8us* p = (const v8us*)gp;
        *(v8us*)&Asb[lrow * LDST + lk]     = p[0];
        *(v8us*)&Asb[lrow * LDST + lk + 8] = p[1];
#endif
      } else {
        const int grs = (gr < M) ? gr : (M - 1);
        #pragma unroll
        for (int j = 0; j < 16; ++j) {
          const int gk  = kk + lk + j;
          const int gks = (gk < K) ? gk : (K - 1);
          unsigned short v =
              TA ? A[(size_t)gks * lda + grs] : A[(size_t)grs * lda + gks];
          if (!(gr < M && gk < K)) v = 0;
          Asb[lrow * LDST + lk + j] = v;
        }
      }
    }
    // B tile transposed: Bsb[n][k] = B(kk+k, bn+n)
    {
      const int gn = bn + lrow;
      if (TB && gn < N && (kk + lk + 16) <= K) {
        const unsigned short* gp = &B[(size_t)gn * ldb + kk + lk];
#if HAS_ASYNC_LDS
        __builtin_amdgcn_global_load_async_to_lds_b128(AS1C(gp),     AS3C(&Bsb[lrow * LDST + lk]),     0, 0);
        __builtin_amdgcn_global_load_async_to_lds_b128(AS1C(gp + 8), AS3C(&Bsb[lrow * LDST + lk + 8]), 0, 0);
#else
        const v8us* p = (const v8us*)gp;
        *(v8us*)&Bsb[lrow * LDST + lk]     = p[0];
        *(v8us*)&Bsb[lrow * LDST + lk + 8] = p[1];
#endif
      } else {
        const int gns = (gn < N) ? gn : (N - 1);
        #pragma unroll
        for (int j = 0; j < 16; ++j) {
          const int gk  = kk + lk + j;
          const int gks = (gk < K) ? gk : (K - 1);
          unsigned short v =
              TB ? B[(size_t)gns * ldb + gks] : B[(size_t)gks * ldb + gns];
          if (!(gn < N && gk < K)) v = 0;
          Bsb[lrow * LDST + lk + j] = v;
        }
      }
    }
  };

  v8f acc[2][4];
  #pragma unroll
  for (int i = 0; i < 2; ++i)
    #pragma unroll
    for (int j = 0; j < 4; ++j)
      acc[i][j] = v8f{0.f,0.f,0.f,0.f,0.f,0.f,0.f,0.f};

  const int nsteps = (K + BK - 1) / BK;

  stage(0, 0);
  wait_async_lds();
  __syncthreads();

  for (int s = 0; s < nsteps; ++s) {
    const int kk  = s * BK;
    const int cur = s & 1;
    if (kk + BK < K) stage(kk + BK, cur ^ 1);   // prefetch next tile (async)

    // ---- fragments per ISA 7.12.2 (16-bit A 16x32; B staged as NxK) -------
    const unsigned short* Asb = &As[cur][0];
    const unsigned short* Bsb = &Bs[cur][0];
    const int khalf = lane >> 4;     // K-half select
    const int l15   = lane & 15;
    v16bf afrag[2], bfrag[4];
    #pragma unroll
    for (int sr = 0; sr < 2; ++sr) {
      const int r = wr * 32 + sr * 16 + l15;
      FragCvt fc;
      fc.p.lo = *(const v8us*)&Asb[r * LDST + khalf * 8];        // K 0-7 / 8-15
      fc.p.hi = *(const v8us*)&Asb[r * LDST + 16 + khalf * 8];   // K 16-23 / 24-31
      afrag[sr] = fc.f;
    }
    #pragma unroll
    for (int sc = 0; sc < 4; ++sc) {
      const int c = wc * 64 + sc * 16 + l15;
      FragCvt fc;
      fc.p.lo = *(const v8us*)&Bsb[c * LDST + khalf * 8];
      fc.p.hi = *(const v8us*)&Bsb[c * LDST + 16 + khalf * 8];
      bfrag[sc] = fc.f;
    }
    #pragma unroll
    for (int sr = 0; sr < 2; ++sr)
      #pragma unroll
      for (int sc = 0; sc < 4; ++sc)
        acc[sr][sc] = __builtin_amdgcn_wmma_f32_16x16x32_bf16(
            false, afrag[sr], false, bfrag[sc],
            (short)0, acc[sr][sc], false, false);

    wait_async_lds();
    __syncthreads();
  }

  // ---- epilogue: C/D layout = lane -> N, VGPR -> M ------------------------
  const int en = lane & 15;
  const int mb = (lane >> 4) * 8;
  #pragma unroll
  for (int sr = 0; sr < 2; ++sr)
    #pragma unroll
    for (int sc = 0; sc < 4; ++sc)
      #pragma unroll
      for (int v = 0; v < 8; ++v) {
        const int m = bm + wr * 32 + sr * 16 + mb + v;
        const int c = bn + wc * 64 + sc * 16 + en;
        if (m < M && c < N) {
          float x = acc[sr][sc][v];
          if (bias) x += bias[c];
          if (EPI == 1) x = fmaxf(x, 0.0f);
          const size_t o = (size_t)m * ldc + c;
          if (Cf)  Cf[o]  = x;
          if (Cbf) Cbf[o] = f2bf(x);
        }
      }
}

// ---------------------------------------------------------------------------
// Elementwise / reduction kernels
// ---------------------------------------------------------------------------
__global__ void k_f32_to_bf16(const float* __restrict__ s,
                              unsigned short* __restrict__ d, size_t n) {
  for (size_t i = (size_t)blockIdx.x * blockDim.x + threadIdx.x; i < n;
       i += (size_t)gridDim.x * blockDim.x)
    d[i] = f2bf(s[i]);
}

// C = (U V^T) * mask, diag -> 0 ; write fp32 (d_out) + bf16 copy
__global__ void k_mask_C(const float* __restrict__ UV,
                         const float* __restrict__ mask,
                         float* __restrict__ Cout,
                         unsigned short* __restrict__ Cbf, int n) {
  const size_t total = (size_t)n * n;
  for (size_t idx = (size_t)blockIdx.x * blockDim.x + threadIdx.x; idx < total;
       idx += (size_t)gridDim.x * blockDim.x) {
    const int i = (int)(idx / n), j = (int)(idx % n);
    float v = (i == j) ? 0.0f : UV[idx] * mask[idx];
    Cout[idx] = v;
    Cbf[idx]  = f2bf(v);
  }
}

// rn[i] = 1 / max(||Z_g[i,:]||, 1e-12)
__global__ void k_rnorm(const float* __restrict__ Z, float* __restrict__ rn,
                        int n, int d) {
  const int i = blockIdx.x * blockDim.x + threadIdx.x;
  if (i >= n) return;
  float ss = 0.f;
  const float* p = Z + (size_t)i * d;
  for (int j = 0; j < d; ++j) ss += p[j] * p[j];
  rn[i] = 1.0f / fmaxf(sqrtf(ss), 1e-12f);
}

// A_pred = sigmoid(G);  G <- cosine sim (in place)
__global__ void k_apred_scos(float* __restrict__ G, float* __restrict__ Apred,
                             const float* __restrict__ rn, int n) {
  const size_t total = (size_t)n * n;
  for (size_t idx = (size_t)blockIdx.x * blockDim.x + threadIdx.x; idx < total;
       idx += (size_t)gridDim.x * blockDim.x) {
    const int i = (int)(idx / n), j = (int)(idx % n);
    const float g = G[idx];
    Apred[idx] = 1.0f / (1.0f + __expf(-g));
    G[idx]     = g * rn[i] * rn[j];
  }
}

// per-row iterative top-k marking (row cached in LDS; deterministic ties)
__global__ __launch_bounds__(256)
void k_topk(const float* __restrict__ S, unsigned char* __restrict__ keep,
            int n, int k) {
  __shared__ float vals[N_];
  __shared__ float smax[256];
  __shared__ int   simx[256];
  const int row = blockIdx.x;
  const int tid = threadIdx.x;
  const float* Sr = S + (size_t)row * n;
  for (int j = tid; j < n; j += 256) {
    vals[j] = Sr[j];
    keep[(size_t)row * n + j] = 0;
  }
  __syncthreads();
  for (int it = 0; it < k; ++it) {
    float m = -3.4e38f; int mi = 0x7fffffff;
    for (int j = tid; j < n; j += 256) {
      const float v = vals[j];
      if (v > m || (v == m && j < mi)) { m = v; mi = j; }
    }
    smax[tid] = m; simx[tid] = mi;
    __syncthreads();
    for (int s = 128; s > 0; s >>= 1) {
      if (tid < s) {
        if (smax[tid + s] > smax[tid] ||
            (smax[tid + s] == smax[tid] && simx[tid + s] < simx[tid])) {
          smax[tid] = smax[tid + s]; simx[tid] = simx[tid + s];
        }
      }
      __syncthreads();
    }
    if (tid == 0) {
      keep[(size_t)row * n + simx[0]] = 1;
      vals[simx[0]] = -3.4e38f;
    }
    __syncthreads();
  }
}

// Sg_un(i,j) = relu(0.5*(keep_ij*S_ij + keep_ji*S_ji)), diag 0
__global__ void k_sg_unnorm(const float* __restrict__ S,
                            const unsigned char* __restrict__ keep,
                            float* __restrict__ Sg, int n) {
  const size_t total = (size_t)n * n;
  for (size_t idx = (size_t)blockIdx.x * blockDim.x + threadIdx.x; idx < total;
       idx += (size_t)gridDim.x * blockDim.x) {
    const int i = (int)(idx / n), j = (int)(idx % n);
    const size_t t = (size_t)j * n + i;
    float a = keep[idx] ? S[idx] : 0.0f;
    float b = keep[t]   ? S[t]   : 0.0f;
    float v = 0.5f * (a + b);
    if (i == j) v = 0.0f;
    Sg[idx] = fmaxf(v, 0.0f);
  }
}

__global__ __launch_bounds__(256)
void k_rowsum(const float* __restrict__ A, float* __restrict__ rs, int n) {
  __shared__ float sm[256];
  const int row = blockIdx.x;
  const float* p = A + (size_t)row * n;
  float s = 0.f;
  for (int j = threadIdx.x; j < n; j += 256) s += p[j];
  sm[threadIdx.x] = s; __syncthreads();
  for (int t = 128; t > 0; t >>= 1) {
    if (threadIdx.x < t) sm[threadIdx.x] += sm[threadIdx.x + t];
    __syncthreads();
  }
  if (threadIdx.x == 0) rs[row] = sm[0];
}

// finalize: Sg normalized (in place), Ss = |C|+|C^T|, S = a*Ss + (1-a)*Sg
__global__ void k_final(float* __restrict__ Sg, const float* __restrict__ rs,
                        const float* __restrict__ C, float* __restrict__ Ss,
                        float* __restrict__ Sout,
                        const float* __restrict__ alpha_raw, int n) {
  const float al = 1.0f / (1.0f + __expf(-alpha_raw[0]));
  const size_t total = (size_t)n * n;
  for (size_t idx = (size_t)blockIdx.x * blockDim.x + threadIdx.x; idx < total;
       idx += (size_t)gridDim.x * blockDim.x) {
    const int i = (int)(idx / n), j = (int)(idx % n);
    const float sg = Sg[idx] / (rs[i] + 1e-8f);
    Sg[idx] = sg;
    const float ss = fabsf(C[idx]) + fabsf(C[(size_t)j * n + i]);
    Ss[idx]   = ss;
    Sout[idx] = al * ss + (1.0f - al) * sg;
  }
}

// recon_s = mean((C^T Z_s - Z_s)^2) ; nuc_s = 0.5*BETA*(sum U^2 + sum V^2)
__global__ __launch_bounds__(256)
void k_scalars(const float* __restrict__ Mrec, const float* __restrict__ Zs,
               const float* __restrict__ U, const float* __restrict__ V,
               float* __restrict__ out_recon, float* __restrict__ out_nuc,
               int nzs, int nur) {
  __shared__ float s1[256], s2[256];
  float a = 0.f, b = 0.f;
  for (int i = threadIdx.x; i < nzs; i += 256) {
    const float d = Mrec[i] - Zs[i]; a += d * d;
  }
  for (int i = threadIdx.x; i < nur; i += 256) {
    const float u = U[i], v = V[i]; b += u * u + v * v;
  }
  s1[threadIdx.x] = a; s2[threadIdx.x] = b; __syncthreads();
  for (int t = 128; t > 0; t >>= 1) {
    if (threadIdx.x < t) {
      s1[threadIdx.x] += s1[threadIdx.x + t];
      s2[threadIdx.x] += s2[threadIdx.x + t];
    }
    __syncthreads();
  }
  if (threadIdx.x == 0) {
    out_recon[0] = s1[0] / (float)nzs;
    out_nuc[0]   = 0.5f * 1.0f * s2[0];
  }
}

// ---------------------------------------------------------------------------
extern "C" void kernel_launch(void* const* d_in, const int* in_sizes, int n_in,
                              void* d_out, int out_size, void* d_ws, size_t ws_size,
                              hipStream_t stream) {
  (void)in_sizes; (void)n_in; (void)out_size; (void)ws_size;
  const float* X      = (const float*)d_in[0];
  const float* A_norm = (const float*)d_in[1];
  const float* mask   = (const float*)d_in[2];
  const float* W1  = (const float*)d_in[3];  const float* b1   = (const float*)d_in[4];
  const float* W2  = (const float*)d_in[5];  const float* b2   = (const float*)d_in[6];
  const float* Wd  = (const float*)d_in[7];  const float* bd   = (const float*)d_in[8];
  const float* We1 = (const float*)d_in[9];  const float* be1  = (const float*)d_in[10];
  const float* We2 = (const float*)d_in[11]; const float* be2  = (const float*)d_in[12];
  const float* U   = (const float*)d_in[13]; const float* V    = (const float*)d_in[14];
  const float* alpha_raw = (const float*)d_in[15];

  const size_t N = N_, D = D_, HID = HID_, ZG = ZG_, ZS = ZS_, R = R_;
  const size_t NN = N * N, ND = N * D;

  // ---- d_out layout (tuple flattened in return order) ----------------------
  float* out        = (float*)d_out;
  float* o_Zg       = out;                       // N*ZG
  float* o_Apred    = o_Zg    + N * ZG;          // N*N
  float* o_Xpred    = o_Apred + NN;              // N*D
  float* o_Zs       = o_Xpred + ND;              // N*ZS
  float* o_C        = o_Zs    + N * ZS;          // N*N
  float* o_Sg       = o_C     + NN;              // N*N
  float* o_Ss       = o_Sg    + NN;              // N*N
  float* o_S        = o_Ss    + NN;              // N*N
  float* o_recon    = o_S     + NN;              // 1
  float* o_nuc      = o_recon + 1;               // 1

  // ---- workspace arena -----------------------------------------------------
  char* ws = (char*)d_ws;
  size_t off = 0;
  auto alloc = [&](size_t bytes) -> char* {
    off = (off + 255) & ~(size_t)255;
    char* p = ws + off; off += bytes; return p;
  };
  unsigned short* Abf   = (unsigned short*)alloc(NN * 2);
  unsigned short* Xbf   = (unsigned short*)alloc(ND * 2);
  unsigned short* AXbf  = (unsigned short*)alloc(ND * 2);
  unsigned short* W1bf  = (unsigned short*)alloc(D * HID * 2);
  unsigned short* hbf   = (unsigned short*)alloc(N * HID * 2);
  unsigned short* Ahbf  = (unsigned short*)alloc(N * HID * 2);
  unsigned short* W2bf  = (unsigned short*)alloc(HID * ZG * 2);
  unsigned short* Zgbf  = (unsigned short*)alloc(N * ZG * 2);
  unsigned short* Wdbf  = (unsigned short*)alloc(ZG * D * 2);
  unsigned short* We1bf = (unsigned short*)alloc(D * HID * 2);
  unsigned short* T1bf  = (unsigned short*)alloc(N * HID * 2);
  unsigned short* We2bf = (unsigned short*)alloc(HID * ZS * 2);
  unsigned short* Zsbf  = (unsigned short*)alloc(N * ZS * 2);
  unsigned short* Ubf   = (unsigned short*)alloc(N * R * 2);
  unsigned short* Vbf   = (unsigned short*)alloc(N * R * 2);
  float*          BIGf  = (float*)alloc(NN * 4);       // UV^T, then G/S_cos
  unsigned short* Cbf   = (unsigned short*)alloc(NN * 2);
  unsigned char*  keep  = (unsigned char*)alloc(NN);
  float*          rn    = (float*)alloc(N * 4);
  float*          rs    = (float*)alloc(N * 4);
  float*          Mrec  = (float*)alloc(N * ZS * 4);

  const int EW_BLOCKS = 8192;
  auto conv = [&](const float* s, unsigned short* d, size_t n) {
    size_t b = (n + 255) / 256; if (b > 8192) b = 8192;
    k_f32_to_bf16<<<dim3((unsigned)b), dim3(256), 0, stream>>>(s, d, n);
  };
  auto ggrid = [](int M, int Nn) { return dim3((Nn + BN - 1) / BN, (M + BM - 1) / BM); };

  // ---- 0) fp32 -> bf16 conversions ----------------------------------------
  conv(A_norm, Abf,  NN);
  conv(X,      Xbf,  ND);
  conv(W1,  W1bf,  D * HID);
  conv(W2,  W2bf,  HID * ZG);
  conv(Wd,  Wdbf,  ZG * D);
  conv(We1, We1bf, D * HID);
  conv(We2, We2bf, HID * ZS);
  conv(U,   Ubf,   N * R);
  conv(V,   Vbf,   N * R);

  // ---- 1) subspace branch --------------------------------------------------
  // T1 = relu(X @ We1 + be1)
  gemm_bf16_wmma<0,0,1><<<ggrid(N, HID), 256, 0, stream>>>(
      Xbf, We1bf, be1, nullptr, T1bf, N, HID, D, D, HID, HID);
  // Z_s = T1 @ We2 + be2  (fp32 to d_out + bf16 copy)
  gemm_bf16_wmma<0,0,0><<<ggrid(N, ZS), 256, 0, stream>>>(
      T1bf, We2bf, be2, o_Zs, Zsbf, N, ZS, HID, HID, ZS, ZS);
  // UV^T (fp32 scratch)
  gemm_bf16_wmma<0,1,0><<<ggrid(N, N), 256, 0, stream>>>(
      Ubf, Vbf, nullptr, BIGf, nullptr, N, N, R, R, R, N);
  // C = (U V^T)*mask, diag 0
  k_mask_C<<<EW_BLOCKS, 256, 0, stream>>>(BIGf, mask, o_C, Cbf, (int)N);
  // Mrec = C^T @ Z_s   (A transposed)
  gemm_bf16_wmma<1,0,0><<<ggrid(N, ZS), 256, 0, stream>>>(
      Cbf, Zsbf, nullptr, Mrec, nullptr, N, ZS, N, N, ZS, ZS);

  // ---- 2) GAE branch -------------------------------------------------------
  // AX = A_norm @ X  (bf16 only)
  gemm_bf16_wmma<0,0,0><<<ggrid(N, D), 256, 0, stream>>>(
      Abf, Xbf, nullptr, nullptr, AXbf, N, D, N, N, D, D);
  // h = relu(AX @ W1 + b1)
  gemm_bf16_wmma<0,0,1><<<ggrid(N, HID), 256, 0, stream>>>(
      AXbf, W1bf, b1, nullptr, hbf, N, HID, D, D, HID, HID);
  // Ah = A_norm @ h
  gemm_bf16_wmma<0,0,0><<<ggrid(N, HID), 256, 0, stream>>>(
      Abf, hbf, nullptr, nullptr, Ahbf, N, HID, N, N, HID, HID);
  // Z_g = relu(Ah @ W2 + b2)
  gemm_bf16_wmma<0,0,1><<<ggrid(N, ZG), 256, 0, stream>>>(
      Ahbf, W2bf, b2, o_Zg, Zgbf, N, ZG, HID, HID, ZG, ZG);
  // G = Z_g @ Z_g^T  (reuse BIGf)
  gemm_bf16_wmma<0,1,0><<<ggrid(N, N), 256, 0, stream>>>(
      Zgbf, Zgbf, nullptr, BIGf, nullptr, N, N, ZG, ZG, ZG, N);
  // X_pred = Z_g @ Wdec + bdec
  gemm_bf16_wmma<0,0,0><<<ggrid(N, D), 256, 0, stream>>>(
      Zgbf, Wdbf, bd, o_Xpred, nullptr, N, D, ZG, ZG, D, D);

  // ---- 3) affinities -------------------------------------------------------
  k_rnorm<<<(unsigned)((N + 255) / 256), 256, 0, stream>>>(o_Zg, rn, (int)N, (int)ZG);
  k_apred_scos<<<EW_BLOCKS, 256, 0, stream>>>(BIGf, o_Apred, rn, (int)N);
  k_topk<<<(unsigned)N, 256, 0, stream>>>(BIGf, keep, (int)N, KNN_);
  k_sg_unnorm<<<EW_BLOCKS, 256, 0, stream>>>(BIGf, keep, o_Sg, (int)N);
  k_rowsum<<<(unsigned)N, 256, 0, stream>>>(o_Sg, rs, (int)N);
  k_final<<<EW_BLOCKS, 256, 0, stream>>>(o_Sg, rs, o_C, o_Ss, o_S, alpha_raw, (int)N);

  // ---- 4) scalar losses ----------------------------------------------------
  k_scalars<<<1, 256, 0, stream>>>(Mrec, o_Zs, U, V, o_recon, o_nuc,
                                   (int)(N * ZS), (int)(N * R));
}
